// SymGPTLanguageModel_22342419874061
// MI455X (gfx1250) — compile-verified, matrix operand order checked by reading, not run on hardware
//
#include <hip/hip_runtime.h>

// ---------------------------------------------------------------------------
// SymGPT forward for MI455X (gfx1250, wave32, WMMA).
// All GEMMs run on v_wmma_f32_16x16x32_bf16 (bf16 in, fp32 accumulate).
// GEMM v2: branchless clamped loads, 32x64 wave tile (8 WMMA / K-step),
// K unrolled x2 (16 WMMA / iteration), global_prefetch on the K stream.
// Workspace layout assumes ws_size >= ~180 MB (buffers are aggressively reused).
// ---------------------------------------------------------------------------

typedef __bf16 bf16;
typedef __attribute__((ext_vector_type(16))) __bf16 v16bf;
typedef __attribute__((ext_vector_type(8)))  float  v8f;

static constexpr int cB = 8, cT = 256, cV = 1024, cK = 4, cE = 1024;
static constexpr int cH = 16, cHS = 64, cDEN = 5, cVOUT = 1024;
static constexpr int cBT = cB * cT;            // 2048
static constexpr int cE4 = 4 * cE;             // 4096
static constexpr int cVK = cV * cK;            // 4096
static constexpr int cKVK = cK * cVK;          // 16384

#define GF_BIAS 1
#define GF_RELU 2
#define GF_RES  4

struct U128 { unsigned int w[4]; };
union Frag16 { v16bf v; U128 q[2]; };

// A-frag: 16(rows) x 32(k) bf16.  Lane l: row = l%16, half = l/16.
// element e of v16bf: A[row, 16*(e/8) + 8*half + (e%8)]  (ISA 7.12.2)
// -> two contiguous 16-byte loads per lane. Bt (N x K) uses the same layout
//    with "row" = n (i.e. B supplied column-major per lane).
__device__ __forceinline__ v16bf load_frag(const bf16* base, int ld) {
  const int lane = threadIdx.x & 31;
  const int r    = lane & 15;
  const int half = lane >> 4;
  const bf16* p = base + (long long)r * ld + half * 8;
  Frag16 f;
  f.q[0] = *(const U128*)(p);
  f.q[1] = *(const U128*)(p + 16);
  return f.v;
}

#define WMMA_BF16(a, b, c) \
  __builtin_amdgcn_wmma_f32_16x16x32_bf16(false, (a), false, (b), (short)0, (c), false, false)

// Generic strided-batched GEMM:  C[m,n] = sum_k A[m,k] * Bt[n,k]  (+bias/relu/res)
// z = blockIdx.z decomposed as (z1 = z/zdiv, z2 = z%zdiv) with dual strides,
// which covers per-(b,head) addressing of (bt, E)-packed Q/K/V/O tensors.
// Requirements (met by every call site): M % 16 == 0, M >= 32, N % 16 == 0,
// N >= 64, Kd % 64 == 0. Loads are clamped in-bounds; stores are guarded.
__global__ __launch_bounds__(256) void wmma_gemm_bf16(
    const bf16* __restrict__ A, const bf16* __restrict__ Bt,
    float* __restrict__ C, const float* __restrict__ bias,
    const float* __restrict__ Res,
    int M, int N, int Kd, int lda, int ldb, int ldc, int zdiv,
    long long sA1, long long sA2, long long sB1, long long sB2,
    long long sC1, long long sC2, int flags)
{
  const int z  = blockIdx.z;
  const int z1 = z / zdiv, z2 = z - z1 * zdiv;
  A  += (long long)z1 * sA1 + (long long)z2 * sA2;
  Bt += (long long)z1 * sB1 + (long long)z2 * sB2;
  const long long coff = (long long)z1 * sC1 + (long long)z2 * sC2;
  C += coff;
  const float* Rp = (flags & GF_RES) ? (Res + coff) : nullptr;

  const int wave = threadIdx.x >> 5;
  const int wm = wave >> 2;                 // 2 waves along M (32 rows each)
  const int wn = wave & 3;                  // 4 waves along N (64 cols each)
  const int m0 = blockIdx.x * 64 + wm * 32;
  const int n0 = blockIdx.y * 256 + wn * 64;

  // Clamped bases: loads always in-bounds; garbage tiles are never stored.
  const int ml = (m0 + 32 <= M) ? m0 : (M - 32);
  const int nl = (n0 + 64 <= N) ? n0 : (N - 64);

  const bf16* Ap0 = A + (long long)ml * lda;
  const bf16* Ap1 = A + (long long)(ml + 16) * lda;
  const bf16* Bp0 = Bt + (long long)nl * ldb;
  const bf16* Bp1 = Bt + (long long)(nl + 16) * ldb;
  const bf16* Bp2 = Bt + (long long)(nl + 32) * ldb;
  const bf16* Bp3 = Bt + (long long)(nl + 48) * ldb;

  v8f acc[2][4] = {};

  for (int k = 0; k < Kd; k += 64) {
    // prefetch the K-stream ~8 steps ahead (speculative; OOB is dropped)
    __builtin_prefetch(Ap0 + k + 256, 0, 1);
    __builtin_prefetch(Ap1 + k + 256, 0, 1);
    __builtin_prefetch(Bp0 + k + 256, 0, 1);
    __builtin_prefetch(Bp1 + k + 256, 0, 1);
    __builtin_prefetch(Bp2 + k + 256, 0, 1);
    __builtin_prefetch(Bp3 + k + 256, 0, 1);

    v16bf a0 = load_frag(Ap0 + k, lda);
    v16bf a1 = load_frag(Ap1 + k, lda);
    v16bf b0 = load_frag(Bp0 + k, ldb);
    v16bf b1 = load_frag(Bp1 + k, ldb);
    v16bf b2 = load_frag(Bp2 + k, ldb);
    v16bf b3 = load_frag(Bp3 + k, ldb);
    acc[0][0] = WMMA_BF16(a0, b0, acc[0][0]);
    acc[0][1] = WMMA_BF16(a0, b1, acc[0][1]);
    acc[0][2] = WMMA_BF16(a0, b2, acc[0][2]);
    acc[0][3] = WMMA_BF16(a0, b3, acc[0][3]);
    acc[1][0] = WMMA_BF16(a1, b0, acc[1][0]);
    acc[1][1] = WMMA_BF16(a1, b1, acc[1][1]);
    acc[1][2] = WMMA_BF16(a1, b2, acc[1][2]);
    acc[1][3] = WMMA_BF16(a1, b3, acc[1][3]);

    a0 = load_frag(Ap0 + k + 32, lda);
    a1 = load_frag(Ap1 + k + 32, lda);
    b0 = load_frag(Bp0 + k + 32, ldb);
    b1 = load_frag(Bp1 + k + 32, ldb);
    b2 = load_frag(Bp2 + k + 32, ldb);
    b3 = load_frag(Bp3 + k + 32, ldb);
    acc[0][0] = WMMA_BF16(a0, b0, acc[0][0]);
    acc[0][1] = WMMA_BF16(a0, b1, acc[0][1]);
    acc[0][2] = WMMA_BF16(a0, b2, acc[0][2]);
    acc[0][3] = WMMA_BF16(a0, b3, acc[0][3]);
    acc[1][0] = WMMA_BF16(a1, b0, acc[1][0]);
    acc[1][1] = WMMA_BF16(a1, b1, acc[1][1]);
    acc[1][2] = WMMA_BF16(a1, b2, acc[1][2]);
    acc[1][3] = WMMA_BF16(a1, b3, acc[1][3]);
  }

  // C/D layout: lane l: col = l%16, VGPR r holds row = r + 8*(l/16)  (ISA 7.12.2)
  const int lane = threadIdx.x & 31;
  const int cn   = lane & 15;
  const int half = lane >> 4;
  const bool mv[2] = { m0 < M, (m0 + 16) < M };
#pragma unroll
  for (int i = 0; i < 2; ++i) {
    if (!mv[i]) continue;
#pragma unroll
    for (int j = 0; j < 4; ++j) {
      const int ncol = n0 + j * 16;
      if (ncol >= N) continue;
      const int col = ncol + cn;
      const float bv = (flags & GF_BIAS) ? bias[col] : 0.f;
#pragma unroll
      for (int r = 0; r < 8; ++r) {
        const int row = m0 + i * 16 + half * 8 + r;
        float v = acc[i][j][r] + bv;
        if (flags & GF_RELU) v = fmaxf(v, 0.f);
        if (flags & GF_RES)  v += Rp[(long long)row * ldc + col];
        C[(long long)row * ldc + col] = v;
      }
    }
  }
}

// dst[z][c*R + r] = (bf16) src[z][r*sld + c]   (batched transpose + convert)
__global__ __launch_bounds__(256) void transpose_f32_to_bf16_kernel(
    const float* __restrict__ src, bf16* __restrict__ dst,
    int R, int Cc, int sld, int zdiv,
    long long sS1, long long sS2, long long sD1, long long sD2, long long n)
{
  const int z = blockIdx.z;
  const int z1 = z / zdiv, z2 = z - z1 * zdiv;
  src += (long long)z1 * sS1 + (long long)z2 * sS2;
  dst += (long long)z1 * sD1 + (long long)z2 * sD2;
  long long i = (long long)blockIdx.x * 256 + threadIdx.x;
  if (i >= n) return;
  const int r = (int)(i / Cc);
  const int c = (int)(i - (long long)r * Cc);
  dst[(long long)c * R + r] = (bf16)src[(long long)r * sld + c];
}

__global__ __launch_bounds__(256) void f32_to_bf16_kernel(
    const float* __restrict__ in, bf16* __restrict__ out, long long n)
{
  long long i = (long long)blockIdx.x * blockDim.x + threadIdx.x;
  const long long stride = (long long)gridDim.x * blockDim.x;
  for (; i < n; i += stride) out[i] = (bf16)in[i];
}

// LayerNorm over ncols, emits bf16 for the following GEMM.
__global__ __launch_bounds__(256) void layernorm_bf16_kernel(
    const float* __restrict__ X, const float* __restrict__ g,
    const float* __restrict__ b, bf16* __restrict__ O, int ncols)
{
  const int row = blockIdx.x;
  const float* x = X + (long long)row * ncols;
  bf16* o = O + (long long)row * ncols;
  const int tid = threadIdx.x;
  float s = 0.f, s2 = 0.f;
  for (int c = tid; c < ncols; c += 256) { float v = x[c]; s += v; s2 += v * v; }
  __shared__ float r1[256], r2[256];
  r1[tid] = s; r2[tid] = s2; __syncthreads();
  for (int st = 128; st > 0; st >>= 1) {
    if (tid < st) { r1[tid] += r1[tid + st]; r2[tid] += r2[tid + st]; }
    __syncthreads();
  }
  const float mean = r1[0] / ncols;
  const float var  = r2[0] / ncols - mean * mean;
  const float inv  = rsqrtf(var + 1e-5f);
  for (int c = tid; c < ncols; c += 256)
    o[c] = (bf16)((x[c] - mean) * inv * g[c] + b[c]);
}

// Causal softmax over rows of (B,H,T,T) fp32 scores -> bf16.
__global__ __launch_bounds__(256) void softmax_causal_kernel(
    const float* __restrict__ S, bf16* __restrict__ O, float scale)
{
  const long long row = blockIdx.x;          // over B*H*T
  const int t = (int)(row & (cT - 1));
  const int s = threadIdx.x;
  const float* sp = S + row * cT;
  const float v = (s <= t) ? sp[s] * scale : -3.0e38f;
  __shared__ float red[256];
  red[s] = v; __syncthreads();
  for (int st = 128; st > 0; st >>= 1) { if (s < st) red[s] = fmaxf(red[s], red[s + st]); __syncthreads(); }
  const float mx = red[0]; __syncthreads();
  const float e = (s <= t) ? __expf(v - mx) : 0.f;
  red[s] = e; __syncthreads();
  for (int st = 128; st > 0; st >>= 1) { if (s < st) red[s] += red[s + st]; __syncthreads(); }
  O[row * cT + s] = (bf16)(e / red[0]);
}

// Symbolic attention: gather cooc[idx_t, idx_s, :4], per-channel softmax over s,
// add identity, emit bf16 att[b][t][s*4+h].
__global__ __launch_bounds__(256) void sym_att_kernel(
    const int* __restrict__ idx, const float* __restrict__ cooc, bf16* __restrict__ att)
{
  const int bt = blockIdx.x;                 // b*T + t
  const int b = bt >> 8, t = bt & 255;
  const int s = threadIdx.x;
  const int it = idx[b * cT + t];
  const int is = idx[b * cT + s];
  const float* c = cooc + ((long long)it * cV + is) * cK;
  float vv[4] = { c[0], c[1], c[2], c[3] };
  float ee[4];
  __shared__ float red[256];
  for (int h = 0; h < 4; ++h) {
    __syncthreads();
    red[s] = vv[h]; __syncthreads();
    for (int st = 128; st > 0; st >>= 1) { if (s < st) red[s] = fmaxf(red[s], red[s + st]); __syncthreads(); }
    const float mx = red[0]; __syncthreads();
    const float e = __expf(vv[h] - mx);
    red[s] = e; __syncthreads();
    for (int st = 128; st > 0; st >>= 1) { if (s < st) red[s] += red[s + st]; __syncthreads(); }
    ee[h] = e / red[0];
  }
  bf16* o = att + ((long long)bt * cT + s) * cK;
  for (int h = 0; h < 4; ++h) o[h] = (bf16)(ee[h] + (s == t ? 1.0f : 0.0f));
}

// Pgt[b][e][s*4+h] = Pbf[idx[b,s]*4 + h][e]   (transposed gather for sa GEMM)
__global__ __launch_bounds__(256) void gather_pgt_kernel(
    const bf16* __restrict__ Pbf, const int* __restrict__ idx, bf16* __restrict__ pgt)
{
  long long i = (long long)blockIdx.x * 256 + threadIdx.x;
  const long long total = (long long)cB * cE * cT * cK;
  if (i >= total) return;
  const int h = (int)(i & 3);
  long long r = i >> 2;
  const int s = (int)(r & (cT - 1)); r >>= 8;
  const int e = (int)(r & (cE - 1));
  const int b = (int)(r >> 10);
  const int v = idx[b * cT + s];
  pgt[i] = Pbf[((long long)v * cK + h) * cE + e];
}

// x[bt][e] += float(idx[bt])
__global__ __launch_bounds__(256) void add_idx_kernel(
    float* __restrict__ x, const int* __restrict__ idx)
{
  long long i = (long long)blockIdx.x * 256 + threadIdx.x;
  const long long total = (long long)cBT * cE;
  if (i >= total) return;
  x[i] += (float)idx[i >> 10];
}

extern "C" void kernel_launch(void* const* d_in, const int* in_sizes, int n_in,
                              void* d_out, int out_size, void* d_ws, size_t ws_size,
                              hipStream_t stream)
{
  (void)in_sizes; (void)n_in; (void)out_size; (void)ws_size;
  const int*   idx  = (const int*)  d_in[0];
  const float* cooc = (const float*)d_in[1];
  const float* spw  = (const float*)d_in[2];
  const float* spb  = (const float*)d_in[3];
  const float* slg  = (const float*)d_in[4];
  const float* slb  = (const float*)d_in[5];
  const float* sfw1 = (const float*)d_in[6];
  const float* sfb1 = (const float*)d_in[7];
  const float* sfw2 = (const float*)d_in[8];
  const float* sfb2 = (const float*)d_in[9];
  const float* wq   = (const float*)d_in[10];
  const float* wk   = (const float*)d_in[11];
  const float* wv   = (const float*)d_in[12];
  const float* apw  = (const float*)d_in[13];
  const float* apb  = (const float*)d_in[14];
  const float* g1   = (const float*)d_in[15];
  const float* b1   = (const float*)d_in[16];
  const float* g2   = (const float*)d_in[17];
  const float* b2   = (const float*)d_in[18];
  const float* fw1  = (const float*)d_in[19];
  const float* fb1  = (const float*)d_in[20];
  const float* fw2  = (const float*)d_in[21];
  const float* fb2  = (const float*)d_in[22];
  const float* lnfg = (const float*)d_in[23];
  const float* lnfb = (const float*)d_in[24];
  const float* lmw  = (const float*)d_in[25];
  const float* lmb  = (const float*)d_in[26];
  float* out = (float*)d_out;

  size_t off = 0;
  auto take = [&](size_t bytes) -> void* {
    void* p = (char*)d_ws + off;
    off += (bytes + 255) & ~(size_t)255;
    return p;
  };
  bf16*  cooc_bf = (bf16*) take((size_t)cV * cV * cK * 2);     // 8 MB
  bf16*  wsym_t  = (bf16*) take((size_t)cE * cKVK * 2);        // 32 MB (reused: scores / ff1 f32)
  float* P32     = (float*)take((size_t)cV * cK * cE * 4);     // 16 MB (reused: att2 bf16)
  bf16*  Pbf     = (bf16*) take((size_t)cV * cK * cE * 2);     // 8 MB
  bf16*  attbf   = (bf16*) take((size_t)cB * cT * cT * cK * 2);// 4 MB
  bf16*  pgt     = (bf16*) take((size_t)cB * cE * cT * cK * 2);// 16 MB (reused: ff1 bf16)
  float* x       = (float*)take((size_t)cBT * cE * 4);         // 8 MB
  bf16*  hbf     = (bf16*) take((size_t)cBT * cE * 2);         // 4 MB
  bf16*  w1t     = (bf16*) take((size_t)cE * cE4 * 2);         // 8 MB
  bf16*  w2t     = (bf16*) take((size_t)cE4 * cE * 2);         // 8 MB
  bf16*  wqt     = (bf16*) take((size_t)cE * cE * 2);
  bf16*  wkt     = (bf16*) take((size_t)cE * cE * 2);
  bf16*  wvt     = (bf16*) take((size_t)cE * cE * 2);
  bf16*  pwt     = (bf16*) take((size_t)cE * cE * 2);
  float* qf      = (float*)take((size_t)cBT * cE * 4);
  float* kf      = (float*)take((size_t)cBT * cE * 4);
  float* vf      = (float*)take((size_t)cBT * cE * 4);
  bf16*  qbf     = (bf16*) take((size_t)cBT * cE * 2);
  bf16*  kbf     = (bf16*) take((size_t)cBT * cE * 2);
  bf16*  vtbf    = (bf16*) take((size_t)cBT * cE * 2);
  float* of      = (float*)take((size_t)cBT * cE * 4);
  bf16*  obf     = (bf16*) take((size_t)cBT * cE * 2);
  bf16*  lmwt    = (bf16*) take((size_t)cVOUT * cE * 2);
  float* scores  = (float*)wsym_t;   // alias (wsym_t dead after P GEMM)
  float* ff1     = (float*)wsym_t;   // alias
  bf16*  att2    = (bf16*) P32;      // alias (P32 dead after convert)
  bf16*  ff1b    = pgt;              // alias (pgt dead after sa GEMM)

  auto cvt = [&](const float* src, bf16* dst, long long n) {
    long long g = (n + 255) / 256; if (g > 65536) g = 65536;
    f32_to_bf16_kernel<<<dim3((unsigned)g), dim3(256), 0, stream>>>(src, dst, n);
  };
  auto tconv = [&](const float* src, bf16* dst, int R, int Cc, int sld, int Z, int zdiv,
                   long long sS1, long long sS2, long long sD1, long long sD2) {
    long long n = (long long)R * Cc;
    dim3 g((unsigned)((n + 255) / 256), 1, (unsigned)Z);
    transpose_f32_to_bf16_kernel<<<g, dim3(256), 0, stream>>>(src, dst, R, Cc, sld, zdiv,
                                                              sS1, sS2, sD1, sD2, n);
  };
  auto gemm = [&](const bf16* A, const bf16* Bt, float* C, const float* bias, const float* Res,
                  int M, int N, int Kd, int lda, int ldb, int ldc, int Z, int zdiv,
                  long long sA1, long long sA2, long long sB1, long long sB2,
                  long long sC1, long long sC2, int flags) {
    dim3 g((unsigned)((M + 63) / 64), (unsigned)((N + 255) / 256), (unsigned)Z);
    wmma_gemm_bf16<<<g, dim3(256), 0, stream>>>(A, Bt, C, bias, Res, M, N, Kd, lda, ldb, ldc,
                                                zdiv, sA1, sA2, sB1, sB2, sC1, sC2, flags);
  };
  auto ln = [&](const float* X, const float* g_, const float* b_, bf16* O) {
    layernorm_bf16_kernel<<<dim3(cBT), dim3(256), 0, stream>>>(X, g_, b_, O, cE);
  };

  // ---- Stage 0: symbolic block (with P-precompute optimization) ----
  cvt(cooc, cooc_bf, (long long)cV * cV * cK);
  tconv(spw, wsym_t, cKVK, cE, cE, 1, 1, 0, 0, 0, 0);          // (16384,E) -> (E,16384)
  // P[v,h,:] = cooc_flat(V x 4096) @ W_h, batched over h
  gemm(cooc_bf, wsym_t, P32, nullptr, nullptr,
       cV, cE, cVK, cVK, cKVK, cK * cE,
       cK, 1, 0, 0, (long long)cVK, 0, (long long)cE, 0, 0);
  cvt(P32, Pbf, (long long)cV * cK * cE);
  sym_att_kernel<<<dim3(cB * cT), dim3(256), 0, stream>>>(idx, cooc, attbf);
  {
    const long long tot = (long long)cB * cE * cT * cK;
    gather_pgt_kernel<<<dim3((unsigned)((tot + 255) / 256)), dim3(256), 0, stream>>>(Pbf, idx, pgt);
  }
  // sa[b,t,:] = att[b,t,:,:] . P[idx[b,:],:,:]  (batched over b) + bias
  gemm(attbf, pgt, x, spb, nullptr,
       cT, cE, cT * cK, cT * cK, cT * cK, cE,
       cB, 1, (long long)cT * cT * cK, 0, (long long)cE * cT * cK, 0,
       (long long)cT * cE, 0, GF_BIAS);
  add_idx_kernel<<<dim3((unsigned)(((long long)cBT * cE + 255) / 256)), dim3(256), 0, stream>>>(x, idx);
  // symbolic FFN
  tconv(sfw1, w1t, cE, cE4, cE4, 1, 1, 0, 0, 0, 0);
  tconv(sfw2, w2t, cE4, cE, cE, 1, 1, 0, 0, 0, 0);
  ln(x, slg, slb, hbf);
  gemm(hbf, w1t, ff1, sfb1, nullptr, cBT, cE4, cE, cE, cE, cE4,
       1, 1, 0, 0, 0, 0, 0, 0, GF_BIAS | GF_RELU);
  cvt(ff1, ff1b, (long long)cBT * cE4);
  gemm(ff1b, w2t, x, sfb2, x, cBT, cE, cE4, cE4, cE4, cE,
       1, 1, 0, 0, 0, 0, 0, 0, GF_BIAS | GF_RES);

  // ---- Stage 1: transformer blocks ----
  for (int l = 0; l < cDEN; ++l) {
    const long long wqkv_off = (long long)l * cH * cE * cHS;
    tconv(wq + wqkv_off, wqt, cE, cHS, cHS, cH, 1, (long long)cE * cHS, 0, (long long)cHS * cE, 0);
    tconv(wk + wqkv_off, wkt, cE, cHS, cHS, cH, 1, (long long)cE * cHS, 0, (long long)cHS * cE, 0);
    tconv(wv + wqkv_off, wvt, cE, cHS, cHS, cH, 1, (long long)cE * cHS, 0, (long long)cHS * cE, 0);
    tconv(apw + (long long)l * cE * cE,  pwt, cE,  cE,  cE,  1, 1, 0, 0, 0, 0);
    tconv(fw1 + (long long)l * cE * cE4, w1t, cE,  cE4, cE4, 1, 1, 0, 0, 0, 0);
    tconv(fw2 + (long long)l * cE4 * cE, w2t, cE4, cE,  cE,  1, 1, 0, 0, 0, 0);

    ln(x, g1 + l * cE, b1 + l * cE, hbf);
    gemm(hbf, wqt, qf, nullptr, nullptr, cBT, cE, cE, cE, cE, cE, 1, 1, 0, 0, 0, 0, 0, 0, 0);
    gemm(hbf, wkt, kf, nullptr, nullptr, cBT, cE, cE, cE, cE, cE, 1, 1, 0, 0, 0, 0, 0, 0, 0);
    gemm(hbf, wvt, vf, nullptr, nullptr, cBT, cE, cE, cE, cE, cE, 1, 1, 0, 0, 0, 0, 0, 0, 0);
    cvt(qf, qbf, (long long)cBT * cE);
    cvt(kf, kbf, (long long)cBT * cE);
    // V -> Vt[b][h][d][t]
    tconv(vf, vtbf, cT, cHS, cE, cB * cH, cH,
          (long long)cT * cE, cHS, (long long)cH * cHS * cT, (long long)cHS * cT);
    // scores[b,h,t,s] = Q . K^T  (batched over b*H)
    gemm(qbf, kbf, scores, nullptr, nullptr,
         cT, cT, cHS, cE, cE, cT,
         cB * cH, cH, (long long)cT * cE, cHS, (long long)cT * cE, cHS,
         (long long)cH * cT * cT, (long long)cT * cT, 0);
    softmax_causal_kernel<<<dim3(cB * cH * cT), dim3(256), 0, stream>>>(scores, att2, 0.125f);
    // o[b,t,h*64+d] = att2 . V
    gemm(att2, vtbf, of, nullptr, nullptr,
         cT, cHS, cT, cT, cT, cE,
         cB * cH, cH, (long long)cH * cT * cT, (long long)cT * cT,
         (long long)cH * cHS * cT, (long long)cHS * cT,
         (long long)cT * cE, cHS, 0);
    cvt(of, obf, (long long)cBT * cE);
    gemm(obf, pwt, x, apb + l * cE, x, cBT, cE, cE, cE, cE, cE,
         1, 1, 0, 0, 0, 0, 0, 0, GF_BIAS | GF_RES);
    ln(x, g2 + l * cE, b2 + l * cE, hbf);
    gemm(hbf, w1t, ff1, fb1 + (long long)l * cE4, nullptr, cBT, cE4, cE, cE, cE, cE4,
         1, 1, 0, 0, 0, 0, 0, 0, GF_BIAS | GF_RELU);
    cvt(ff1, ff1b, (long long)cBT * cE4);
    gemm(ff1b, w2t, x, fb2 + l * cE, x, cBT, cE, cE4, cE4, cE4, cE,
         1, 1, 0, 0, 0, 0, 0, 0, GF_BIAS | GF_RES);
  }

  // ---- Stage 2: final LN + LM head ----
  ln(x, lnfg, lnfb, hbf);
  tconv(lmw, lmwt, cE, cVOUT, cVOUT, 1, 1, 0, 0, 0, 0);
  gemm(hbf, lmwt, out, lmb, nullptr, cBT, cVOUT, cE, cE, cE, cVOUT,
       1, 1, 0, 0, 0, 0, 0, 0, GF_BIAS);
}
